// GroupedLinear_35364760715975
// MI455X (gfx1250) — matrix operand order repaired and checked by reading, not compiled, not run on hardware
//
#include <hip/hip_runtime.h>

// Grouped (block-diagonal) linear for MI455X / gfx1250.
// fp32 in/out; compute via split-bf16 (hi+lo) on v_wmma_f32_16x16x32_bf16.
// Roofline: ~268 MB HBM traffic -> ~11.5us floor @23.3TB/s. W (4 MiB) is
// pre-split to bf16 hi/lo planes in d_ws once per launch (L2-resident),
// removing per-block re-conversion VALU; B tiles then move global->LDS via
// gfx1250 async-LDS loads (ASYNCcnt), bypassing VGPRs.

#define BATCHN      8192
#define NGROUP      16
#define GIN         256
#define GOUT        256
#define FEAT        4096
#define MT          64          // batch rows per block
#define KC          64          // K chunk staged in LDS
#define KP          72          // padded LDS K-stride (bf16 elems): 144B rows
#define WELEMS      (NGROUP * GOUT * GIN)   // 1,048,576
#define WS_NEEDED   ((size_t)2 * WELEMS * 2)  // hi+lo bf16 planes = 4 MiB

typedef __bf16 bf16;
typedef __attribute__((ext_vector_type(16))) bf16  v16bf;
typedef __attribute__((ext_vector_type(8)))  bf16  v8bf;
typedef __attribute__((ext_vector_type(4)))  bf16  v4bf;
typedef __attribute__((ext_vector_type(8)))  float v8f;

union V16U { v16bf v; v8bf h[2]; };

static __device__ __forceinline__ bf16 f32_to_bf16_rne(float f) {
    unsigned u = __float_as_uint(f);
    u += 0x7FFFu + ((u >> 16) & 1u);            // round-to-nearest-even
    unsigned short s = (unsigned short)(u >> 16);
    return __builtin_bit_cast(bf16, s);
}
static __device__ __forceinline__ float bf16_to_f32(bf16 h) {
    unsigned short s = __builtin_bit_cast(unsigned short, h);
    return __uint_as_float(((unsigned)s) << 16);
}
static __device__ __forceinline__ void split4(float4 v, v4bf& hv, v4bf& lv) {
    float xs0 = v.x, xs1 = v.y, xs2 = v.z, xs3 = v.w;
    bf16 h0 = f32_to_bf16_rne(xs0); hv[0] = h0; lv[0] = f32_to_bf16_rne(xs0 - bf16_to_f32(h0));
    bf16 h1 = f32_to_bf16_rne(xs1); hv[1] = h1; lv[1] = f32_to_bf16_rne(xs1 - bf16_to_f32(h1));
    bf16 h2 = f32_to_bf16_rne(xs2); hv[2] = h2; lv[2] = f32_to_bf16_rne(xs2 - bf16_to_f32(h2));
    bf16 h3 = f32_to_bf16_rne(xs3); hv[3] = h3; lv[3] = f32_to_bf16_rne(xs3 - bf16_to_f32(h3));
}

// gfx1250 async global->LDS copy (16B), tracked by ASYNCcnt. lds_addr is the
// raw 32-bit LDS byte offset (low 32 bits of a generic shared pointer).
static __device__ __forceinline__ void async_load_b128(unsigned lds_addr, const void* gptr) {
    asm volatile("global_load_async_to_lds_b128 %0, %1, off"
                 :: "v"(lds_addr), "v"((unsigned long long)(uintptr_t)gptr)
                 : "memory");
}
static __device__ __forceinline__ void wait_async0() {
    asm volatile("s_wait_asynccnt 0x0" ::: "memory");
}

// ---------------- prepass: W fp32 -> bf16 hi/lo planes in d_ws ----------------
__global__ __launch_bounds__(256)
void convert_W_split(const float* __restrict__ W, bf16* __restrict__ whi, bf16* __restrict__ wlo) {
    const size_t i = ((size_t)blockIdx.x * 256 + threadIdx.x) * 4;
    float4 v = *(const float4*)(W + i);
    v4bf hv, lv; split4(v, hv, lv);
    *(v4bf*)(whi + i) = hv;
    *(v4bf*)(wlo + i) = lv;
}

// ---------------- main kernel (pre-split W path) ----------------
__global__ __launch_bounds__(256)
void grouped_linear_wmma_ws(const float* __restrict__ x,
                            const bf16* __restrict__ whi,
                            const bf16* __restrict__ wlo,
                            const float* __restrict__ bias,
                            float* __restrict__ y) {
    __shared__ __align__(16) bf16 sA[2 * MT   * KP];   // 18 KB: x hi|lo
    __shared__ __align__(16) bf16 sB[2 * GOUT * KP];   // 72 KB: W hi|lo

    const int tid  = threadIdx.x;
    const int lane = tid & 31;
    const int wave = tid >> 5;
    const int l15  = lane & 15;
    const int lhi  = (lane >> 4) & 1;

    const int m0 = blockIdx.x * MT;
    const int g  = blockIdx.y;

    const float* xg = x + (size_t)m0 * FEAT + g * GIN;
    const bf16*  Wp[2] = { whi + (size_t)g * GOUT * GIN,
                           wlo + (size_t)g * GOUT * GIN };

    v8f acc[4][2];
    #pragma unroll
    for (int mt = 0; mt < 4; ++mt)
        #pragma unroll
        for (int nt = 0; nt < 2; ++nt)
            acc[mt][nt] = (v8f){0.f, 0.f, 0.f, 0.f, 0.f, 0.f, 0.f, 0.f};

    for (int kc = 0; kc < GIN; kc += KC) {
        __syncthreads();

        // ---- stage A: x tile 64x64 fp32 -> bf16 hi/lo in LDS (VALU split) ----
        {
            const int r = tid >> 2;               // 0..63
            const int q = tid & 3;
            const float4* src = (const float4*)(xg + (size_t)r * FEAT + kc);
            #pragma unroll
            for (int i = 0; i < 4; ++i) {
                float4 v = src[q + i * 4];
                const int col = (q + i * 4) * 4;
                v4bf hv, lv; split4(v, hv, lv);
                *(v4bf*)&sA[r * KP + col]           = hv;
                *(v4bf*)&sA[MT * KP + r * KP + col] = lv;
            }
        }
        // ---- stage B: pre-split bf16 W tiles, async global->LDS (no VGPRs) ----
        {
            const int h = tid & 1;                // half-row: 64B each
            #pragma unroll
            for (int pl = 0; pl < 2; ++pl) {      // hi plane, lo plane
                #pragma unroll
                for (int p = 0; p < 2; ++p) {     // row passes: 128 rows each
                    const int r = p * 128 + (tid >> 1);
                    const bf16* grow = Wp[pl] + (size_t)r * GIN + kc;
                    const unsigned lrow =
                        (unsigned)(uintptr_t)&sB[pl * GOUT * KP + r * KP];
                    #pragma unroll
                    for (int i = 0; i < 4; ++i) {
                        const int col = h * 32 + i * 8;      // bf16 elems, 16B step
                        async_load_b128(lrow + col * 2, grow + col);
                    }
                }
            }
        }
        wait_async0();
        __syncthreads();

        // ---- 2 K-steps of 32 per chunk ----
        #pragma unroll
        for (int ks = 0; ks < 2; ++ks) {
            v16bf bh[2], bl[2];
            #pragma unroll
            for (int nt = 0; nt < 2; ++nt) {
                const int n  = wave * 32 + nt * 16 + l15;
                const int kb = ks * 32 + lhi * 16;
                const bf16* ph = &sB[n * KP + kb];
                const bf16* pl = &sB[GOUT * KP + n * KP + kb];
                V16U t;
                t.h[0] = *(const v8bf*)ph;       t.h[1] = *(const v8bf*)(ph + 8);
                bh[nt] = t.v;
                t.h[0] = *(const v8bf*)pl;       t.h[1] = *(const v8bf*)(pl + 8);
                bl[nt] = t.v;
            }
            #pragma unroll
            for (int mt = 0; mt < 4; ++mt) {
                const int m  = mt * 16 + l15;
                const int kb = ks * 32 + lhi * 16;
                const bf16* ph = &sA[m * KP + kb];
                const bf16* pl = &sA[MT * KP + m * KP + kb];
                V16U t;
                t.h[0] = *(const v8bf*)ph;       t.h[1] = *(const v8bf*)(ph + 8);
                const v16bf ah = t.v;
                t.h[0] = *(const v8bf*)pl;       t.h[1] = *(const v8bf*)(pl + 8);
                const v16bf al = t.v;
                #pragma unroll
                for (int nt = 0; nt < 2; ++nt) {
                    acc[mt][nt] = __builtin_amdgcn_wmma_f32_16x16x32_bf16(
                        false, ah, false, bh[nt], (short)0, acc[mt][nt], false, false);
                    acc[mt][nt] = __builtin_amdgcn_wmma_f32_16x16x32_bf16(
                        false, ah, false, bl[nt], (short)0, acc[mt][nt], false, false);
                    acc[mt][nt] = __builtin_amdgcn_wmma_f32_16x16x32_bf16(
                        false, al, false, bh[nt], (short)0, acc[mt][nt], false, false);
                }
            }
        }
    }

    #pragma unroll
    for (int nt = 0; nt < 2; ++nt) {
        const int coln = wave * 32 + nt * 16 + l15;
        const float bv = bias[g * GOUT + coln];
        const int colg = g * GOUT + coln;
        #pragma unroll
        for (int mt = 0; mt < 4; ++mt) {
            const int row = m0 + mt * 16 + lhi * 8;
            #pragma unroll
            for (int r = 0; r < 8; ++r)
                y[(size_t)(row + r) * FEAT + colg] = acc[mt][nt][r] + bv;
        }
    }
}

// ---------------- fallback (round-1 kernel, in-kernel W conversion) ----------------
__global__ __launch_bounds__(256)
void grouped_linear_wmma_fb(const float* __restrict__ x,
                            const float* __restrict__ W,
                            const float* __restrict__ bias,
                            float* __restrict__ y) {
    __shared__ __align__(16) bf16 sA[2 * MT   * KP];
    __shared__ __align__(16) bf16 sB[2 * GOUT * KP];

    const int tid  = threadIdx.x;
    const int lane = tid & 31;
    const int wave = tid >> 5;
    const int l15  = lane & 15;
    const int lhi  = (lane >> 4) & 1;
    const int m0   = blockIdx.x * MT;
    const int g    = blockIdx.y;

    const float* xg = x + (size_t)m0 * FEAT + g * GIN;
    const float* Wg = W + (size_t)g * GOUT * GIN;

    v8f acc[4][2];
    #pragma unroll
    for (int mt = 0; mt < 4; ++mt)
        #pragma unroll
        for (int nt = 0; nt < 2; ++nt)
            acc[mt][nt] = (v8f){0.f, 0.f, 0.f, 0.f, 0.f, 0.f, 0.f, 0.f};

    for (int kc = 0; kc < GIN; kc += KC) {
        __syncthreads();
        {
            const int r = tid >> 2, q = tid & 3;
            const float4* src = (const float4*)(xg + (size_t)r * FEAT + kc);
            #pragma unroll
            for (int i = 0; i < 4; ++i) {
                float4 v = src[q + i * 4];
                const int col = (q + i * 4) * 4;
                v4bf hv, lv; split4(v, hv, lv);
                *(v4bf*)&sA[r * KP + col]           = hv;
                *(v4bf*)&sA[MT * KP + r * KP + col] = lv;
            }
        }
        #pragma unroll
        for (int p = 0; p < 4; ++p) {
            const int r = p * 64 + (tid >> 2), q = tid & 3;
            const float4* src = (const float4*)(Wg + (size_t)r * GIN + kc);
            #pragma unroll
            for (int i = 0; i < 4; ++i) {
                float4 v = src[q + i * 4];
                const int col = (q + i * 4) * 4;
                v4bf hv, lv; split4(v, hv, lv);
                *(v4bf*)&sB[r * KP + col]             = hv;
                *(v4bf*)&sB[GOUT * KP + r * KP + col] = lv;
            }
        }
        __syncthreads();

        #pragma unroll
        for (int ks = 0; ks < 2; ++ks) {
            v16bf bh[2], bl[2];
            #pragma unroll
            for (int nt = 0; nt < 2; ++nt) {
                const int n  = wave * 32 + nt * 16 + l15;
                const int kb = ks * 32 + lhi * 16;
                const bf16* ph = &sB[n * KP + kb];
                const bf16* pl = &sB[GOUT * KP + n * KP + kb];
                V16U t;
                t.h[0] = *(const v8bf*)ph;  t.h[1] = *(const v8bf*)(ph + 8);
                bh[nt] = t.v;
                t.h[0] = *(const v8bf*)pl;  t.h[1] = *(const v8bf*)(pl + 8);
                bl[nt] = t.v;
            }
            #pragma unroll
            for (int mt = 0; mt < 4; ++mt) {
                const int m  = mt * 16 + l15;
                const int kb = ks * 32 + lhi * 16;
                const bf16* ph = &sA[m * KP + kb];
                const bf16* pl = &sA[MT * KP + m * KP + kb];
                V16U t;
                t.h[0] = *(const v8bf*)ph;  t.h[1] = *(const v8bf*)(ph + 8);
                const v16bf ah = t.v;
                t.h[0] = *(const v8bf*)pl;  t.h[1] = *(const v8bf*)(pl + 8);
                const v16bf al = t.v;
                #pragma unroll
                for (int nt = 0; nt < 2; ++nt) {
                    acc[mt][nt] = __builtin_amdgcn_wmma_f32_16x16x32_bf16(
                        false, ah, false, bh[nt], (short)0, acc[mt][nt], false, false);
                    acc[mt][nt] = __builtin_amdgcn_wmma_f32_16x16x32_bf16(
                        false, ah, false, bl[nt], (short)0, acc[mt][nt], false, false);
                    acc[mt][nt] = __builtin_amdgcn_wmma_f32_16x16x32_bf16(
                        false, al, false, bh[nt], (short)0, acc[mt][nt], false, false);
                }
            }
        }
    }

    #pragma unroll
    for (int nt = 0; nt < 2; ++nt) {
        const int coln = wave * 32 + nt * 16 + l15;
        const float bv = bias[g * GOUT + coln];
        const int colg = g * GOUT + coln;
        #pragma unroll
        for (int mt = 0; mt < 4; ++mt) {
            const int row = m0 + mt * 16 + lhi * 8;
            #pragma unroll
            for (int r = 0; r < 8; ++r)
                y[(size_t)(row + r) * FEAT + colg] = acc[mt][nt][r] + bv;
        }
    }
}

extern "C" void kernel_launch(void* const* d_in, const int* in_sizes, int n_in,
                              void* d_out, int out_size, void* d_ws, size_t ws_size,
                              hipStream_t stream) {
    (void)in_sizes; (void)n_in; (void)out_size;
    const float* x    = (const float*)d_in[0];
    const float* W    = (const float*)d_in[1];
    const float* bias = (const float*)d_in[2];
    float*       y    = (float*)d_out;
    dim3 grid(BATCHN / MT, NGROUP);

    if (ws_size >= WS_NEEDED) {
        bf16* whi = (bf16*)d_ws;
        bf16* wlo = whi + WELEMS;
        convert_W_split<<<dim3(WELEMS / (256 * 4)), dim3(256), 0, stream>>>(W, whi, wlo);
        grouped_linear_wmma_ws<<<grid, dim3(256), 0, stream>>>(x, whi, wlo, bias, y);
    } else {
        grouped_linear_wmma_fb<<<grid, dim3(256), 0, stream>>>(x, W, bias, y);
    }
}
